// Model_45122926411814
// MI455X (gfx1250) — compile-verified, matrix operand order checked by reading, not run on hardware
//
#include <hip/hip_runtime.h>
#include <hip/hip_bf16.h>

// ---- problem constants (match reference) ----
#define NN   100000   // nodes
#define EE   400000   // edges
#define PP   10000    // fragments
#define FE2  20000    // fragment edges (before symmetrization)
#define EMB  300
#define HID  600
#define LAYERS 5
#define PAD_E 384     // padded 300 (multiple of 128)
#define PAD_H 640     // padded 600 (multiple of 128)

typedef __attribute__((ext_vector_type(16))) __bf16 v16bf;
typedef __attribute__((ext_vector_type(8)))  float  v8f;

// =====================================================================
// bf16 WMMA GEMM:  C[M x Ncols] = act(A[M x K] * B[K x Ncols] + bias)
// A : bf16 row-major, row length = Kpad (zero padded past valid K)
// Bt: bf16 transposed+padded: Bt[n * Kpad + k]  (zeros in padding)
// biasP: f32 padded to Npad (zeros past Ncols)
// Block: 256 threads = 8 waves; tile 128(M) x 128(N); K step 32.
// Wave w: rows [32*(w&3), +32), cols [64*(w>>2), +64)
//   -> 2 A fragments x 4 B fragments = 8 WMMAs / wave / K-step.
// Software pipelined: global loads for tile k+1 are issued before the
// WMMA chain of tile k and only waited at the LDS store.
// =====================================================================
template<bool RELU, bool ST_F32, bool ST_BF16>
__global__ __launch_bounds__(256)
void gemm_wmma_bf16(const __bf16* __restrict__ A, int Kpad,
                    const __bf16* __restrict__ Bt,
                    const float* __restrict__ biasP,
                    int M, int Ncols,
                    float* __restrict__ Cf,
                    __bf16* __restrict__ Cb, int ldcb)
{
    __shared__ __align__(16) __bf16 As[128][32];
    __shared__ __align__(16) __bf16 Bs[128][32];

    const int mBase = blockIdx.x * 128;
    const int nBase = blockIdx.y * 128;
    const int tid   = threadIdx.x;
    const int wave  = tid >> 5;
    const int lane  = tid & 31;
    const int waveM = wave & 3;        // 4 row groups of 32
    const int waveN = wave >> 2;       // 2 col groups of 64

    v8f acc[2][4] = {};

    // staging coords (both tiles are 128 x 32 bf16 = 8KB, 32B per thread)
    const int sRow   = tid >> 1;
    const int sKoff  = (tid & 1) * 16;
    const bool aValid = (mBase + sRow) < M;
    const __bf16* aSrc = A  + (size_t)(mBase + sRow) * Kpad + sKoff;
    const __bf16* bSrc = Bt + (size_t)(nBase + sRow) * Kpad + sKoff;

    uint4 pA0, pA1, pB0, pB1;
    auto loadTiles = [&](int k0) {
        uint4 z0 = {0,0,0,0}, z1 = {0,0,0,0};
        if (aValid) {
            const uint4* p = (const uint4*)(aSrc + k0);
            z0 = p[0]; z1 = p[1];
        }
        pA0 = z0; pA1 = z1;
        const uint4* q = (const uint4*)(bSrc + k0);
        pB0 = q[0]; pB1 = q[1];
    };
    auto storeTiles = [&]() {
        *(uint4*)&As[sRow][sKoff]     = pA0;
        *(uint4*)&As[sRow][sKoff + 8] = pA1;
        *(uint4*)&Bs[sRow][sKoff]     = pB0;
        *(uint4*)&Bs[sRow][sKoff + 8] = pB1;
    };

    const int nIter = Kpad >> 5;
    loadTiles(0);
    storeTiles();
    __syncthreads();

    const int ksel  = (lane >> 4) * 8;
    const int mrow0 = waveM * 32 + (lane & 15);
    const int ncol  = waveN * 64 + (lane & 15);

    for (int it = 0; it < nIter; ++it) {
        // ---- prefetch next tile (latency hidden behind WMMA chain) ----
        if (it + 1 < nIter) loadTiles((it + 1) << 5);

        // ---- fragments (CDNA5 16-bit layout: lane L -> row/col L%16;
        //      lanes 0-15: K[0..7],[16..23]; lanes 16-31: K[8..15],[24..31]) ----
        union Frag { v16bf v; uint4 q[2]; };
        Frag a0, a1, b[4];
        a0.q[0] = *(const uint4*)&As[mrow0][ksel];
        a0.q[1] = *(const uint4*)&As[mrow0][ksel + 16];
        a1.q[0] = *(const uint4*)&As[mrow0 + 16][ksel];
        a1.q[1] = *(const uint4*)&As[mrow0 + 16][ksel + 16];
        #pragma unroll
        for (int t = 0; t < 4; ++t) {
            b[t].q[0] = *(const uint4*)&Bs[ncol + t * 16][ksel];
            b[t].q[1] = *(const uint4*)&Bs[ncol + t * 16][ksel + 16];
        }
        #pragma unroll
        for (int t = 0; t < 4; ++t)
            acc[0][t] = __builtin_amdgcn_wmma_f32_16x16x32_bf16(
                false, a0.v, false, b[t].v, (short)0, acc[0][t], false, false);
        #pragma unroll
        for (int t = 0; t < 4; ++t)
            acc[1][t] = __builtin_amdgcn_wmma_f32_16x16x32_bf16(
                false, a1.v, false, b[t].v, (short)0, acc[1][t], false, false);

        if (it + 1 < nIter) {
            __syncthreads();     // all waves done reading this tile
            storeTiles();        // waits on loadcnt here, after WMMAs
            __syncthreads();
        }
    }

    // ---- epilogue: C tile layout lane L, vgpr j -> m = 8*(L/16)+j, n = L%16 ----
    #pragma unroll
    for (int i = 0; i < 2; ++i) {
        #pragma unroll
        for (int t = 0; t < 4; ++t) {
            const int n = nBase + waveN * 64 + t * 16 + (lane & 15);
            const float bia = biasP[n];
            #pragma unroll
            for (int j = 0; j < 8; ++j) {
                const int m = mBase + waveM * 32 + i * 16 + (lane >> 4) * 8 + j;
                float v = acc[i][t][j] + bia;
                if (RELU) v = v > 0.0f ? v : 0.0f;
                if (m < M) {
                    if (ST_F32 && n < Ncols) Cf[(size_t)m * Ncols + n] = v;
                    if (ST_BF16)             Cb[(size_t)m * ldcb + n] = (__bf16)v;
                }
            }
        }
    }
}

// ===================== f32 -> zero-padded bf16 (pairwise, vectorized) ==========
__global__ void k_cvt_pad(const float* __restrict__ src, __bf16* __restrict__ dst,
                          int M, int Kvalid, int Kpad) {
    const int pairsPerRow = Kpad >> 1;
    size_t idx = (size_t)blockIdx.x * blockDim.x + threadIdx.x;
    if (idx >= (size_t)M * pairsPerRow) return;
    int row = (int)(idx / pairsPerRow);
    int c   = ((int)(idx % pairsPerRow)) << 1;
    float2 v = make_float2(0.0f, 0.0f);
    if (c + 1 < Kvalid) v = *(const float2*)(src + (size_t)row * Kvalid + c);
    union { __bf16 h[2]; unsigned u; } p;
    p.h[0] = (__bf16)v.x;
    p.h[1] = (__bf16)v.y;
    ((unsigned*)dst)[idx] = p.u;
}

// ===================== weight / bias prep =====================
__global__ void k_conv_wT(const float* __restrict__ W, __bf16* __restrict__ Bt,
                          int Kvalid, int Ncol, int Kpad, int Npad) {
    int idx = blockIdx.x * blockDim.x + threadIdx.x;
    if (idx >= Npad * Kpad) return;
    int n = idx / Kpad, k = idx % Kpad;
    float v = (n < Ncol && k < Kvalid) ? W[(size_t)k * Ncol + n] : 0.0f;
    Bt[idx] = (__bf16)v;
}

__global__ void k_pad_bias(const float* __restrict__ b, float* __restrict__ bp,
                           int Ncol, int Npad) {
    int n = blockIdx.x * blockDim.x + threadIdx.x;
    if (n >= Npad) return;
    bp[n] = (n < Ncol) ? b[n] : 0.0f;
}

// ===================== GNN elementwise / scatter =====================
__global__ void k_embed(const int* __restrict__ x,
                        const float* __restrict__ e1, const float* __restrict__ e2,
                        float* __restrict__ h) {
    size_t i = (size_t)blockIdx.x * blockDim.x + threadIdx.x;
    if (i >= (size_t)NN * EMB) return;
    int n = (int)(i / EMB), c = (int)(i % EMB);
    h[i] = e1[(size_t)x[n * 2] * EMB + c] + e2[(size_t)x[n * 2 + 1] * EMB + c];
}

__global__ void k_agg_init(const float* __restrict__ h,
                           const float* __restrict__ ge1, const float* __restrict__ ge2,
                           float* __restrict__ agg) {
    size_t i = (size_t)blockIdx.x * blockDim.x + threadIdx.x;
    if (i >= (size_t)NN * EMB) return;
    int c = (int)(i % EMB);
    agg[i] = h[i] + ge1[4 * EMB + c] + ge2[c];   // self loop: type 4, dir 0
}

__global__ void k_scatter(const float* __restrict__ h, const int* __restrict__ ei,
                          const int* __restrict__ ea,
                          const float* __restrict__ ge1, const float* __restrict__ ge2,
                          float* __restrict__ agg) {
    size_t i = (size_t)blockIdx.x * blockDim.x + threadIdx.x;
    if (i >= (size_t)EE * EMB) return;
    int e = (int)(i / EMB), c = (int)(i % EMB);
    int s = ei[e], d = ei[EE + e];
    int a0 = ea[e * 2], a1 = ea[e * 2 + 1];
    float v = h[(size_t)s * EMB + c] + ge1[(size_t)a0 * EMB + c]
                                     + ge2[(size_t)a1 * EMB + c];
    atomicAdd(&agg[(size_t)d * EMB + c], v);
}

// ===================== batch norm =====================
__global__ void k_bn_stats(const float* __restrict__ h2, float* __restrict__ stats) {
    int t = threadIdx.x;
    int chunk = (NN + gridDim.x - 1) / gridDim.x;
    int r0 = blockIdx.x * chunk;
    int r1 = r0 + chunk; if (r1 > NN) r1 = NN;
    float s0 = 0.f, q0 = 0.f, s1 = 0.f, q1 = 0.f;
    const int c0 = t, c1 = t + 256;
    for (int r = r0; r < r1; ++r) {
        const float* row = h2 + (size_t)r * EMB;
        if (c0 < EMB) { float v = row[c0]; s0 += v; q0 += v * v; }
        if (c1 < EMB) { float v = row[c1]; s1 += v; q1 += v * v; }
    }
    if (c0 < EMB) { atomicAdd(&stats[c0], s0); atomicAdd(&stats[EMB + c0], q0); }
    if (c1 < EMB) { atomicAdd(&stats[c1], s1); atomicAdd(&stats[EMB + c1], q1); }
}

__global__ void k_bn_fin(const float* __restrict__ stats,
                         const float* __restrict__ g, const float* __restrict__ b,
                         float* __restrict__ ss) {
    int c = blockIdx.x * blockDim.x + threadIdx.x;
    if (c >= EMB) return;
    float mu  = stats[c] * (1.0f / NN);
    float var = stats[EMB + c] * (1.0f / NN) - mu * mu;
    float sc  = g[c] * rsqrtf(var + 1e-5f);
    ss[c] = sc;
    ss[EMB + c] = b[c] - mu * sc;
}

__global__ void k_bn_apply(const float* __restrict__ h2, const float* __restrict__ ss,
                           float* __restrict__ h, int relu) {
    size_t i = (size_t)blockIdx.x * blockDim.x + threadIdx.x;
    if (i >= (size_t)NN * EMB) return;
    int c = (int)(i % EMB);
    float v = h2[i] * ss[c] + ss[EMB + c];
    if (relu) v = fmaxf(v, 0.0f);
    h[i] = v;
}

// ===================== pooling =====================
__global__ void k_pool(const float* __restrict__ h, const int* __restrict__ fb,
                       float* __restrict__ seg, float* __restrict__ cnt) {
    size_t i = (size_t)blockIdx.x * blockDim.x + threadIdx.x;
    if (i >= (size_t)NN * EMB) return;
    int n = (int)(i / EMB), c = (int)(i % EMB);
    int p = fb[n];
    atomicAdd(&seg[(size_t)p * EMB + c], h[i]);
    if (c == 0) atomicAdd(&cnt[p], 1.0f);
}

__global__ void k_pool_div(const float* __restrict__ seg, const float* __restrict__ cnt,
                           __bf16* __restrict__ pb) {
    int i = blockIdx.x * blockDim.x + threadIdx.x;
    if (i >= PP * PAD_E) return;
    int p = i / PAD_E, c = i % PAD_E;
    float v = 0.0f;
    if (c < EMB) v = seg[(size_t)p * EMB + c] / fmaxf(cnt[p], 1.0f);
    pb[i] = (__bf16)v;
}

// ===================== row L2 normalize (wave per row) =====================
__global__ void k_rownorm(const float* __restrict__ X, float* __restrict__ Y, int rows) {
    int wave = threadIdx.x >> 5, lane = threadIdx.x & 31;
    int r = blockIdx.x * 8 + wave;
    if (r >= rows) return;
    const float* x = X + (size_t)r * EMB;
    float s = 0.f;
    for (int c = lane; c < EMB; c += 32) { float v = x[c]; s += v * v; }
    #pragma unroll
    for (int m = 16; m; m >>= 1) s += __shfl_xor(s, m, 32);
    float inv = 1.0f / fmaxf(sqrtf(s), 1e-12f);
    float* y = Y + (size_t)r * EMB;
    for (int c = lane; c < EMB; c += 32) y[c] = x[c] * inv;
}

// ===================== GCN predictor scatter =====================
__global__ void k_deg(const int* __restrict__ fei, float* __restrict__ deg) {
    int i = blockIdx.x * blockDim.x + threadIdx.x;
    if (i >= 2 * FE2) return;
    atomicAdd(&deg[fei[i]], 1.0f);   // u-list == fei flat (concat(row0,row1))
}

__global__ void k_dis(const float* __restrict__ deg, float* __restrict__ dis) {
    int p = blockIdx.x * blockDim.x + threadIdx.x;
    if (p >= PP) return;
    float d = deg[p];
    dis[p] = d > 0.0f ? rsqrtf(fmaxf(d, 1.0f)) : 0.0f;
}

__global__ void k_gcn_scatter(const float* __restrict__ xw, const int* __restrict__ fei,
                              const int* __restrict__ ea,
                              const float* __restrict__ ge1, const float* __restrict__ ge2,
                              const float* __restrict__ dis, float* __restrict__ pred) {
    size_t idx = (size_t)blockIdx.x * blockDim.x + threadIdx.x;
    if (idx >= (size_t)2 * FE2 * EMB) return;
    int i = (int)(idx / EMB), c = (int)(idx % EMB);
    int u = fei[i];
    int v = (i < FE2) ? fei[FE2 + i] : fei[i - FE2];
    int j = (i < FE2) ? i : i - FE2;
    int a0 = ea[j * 2], a1 = ea[j * 2 + 1];
    float nrm = dis[u] * dis[v];
    float val = nrm * (xw[(size_t)u * EMB + c] + ge1[(size_t)a0 * EMB + c]
                                               + ge2[(size_t)a1 * EMB + c]);
    atomicAdd(&pred[(size_t)v * EMB + c], val);
}

// ===================== classifier (wave per row) =====================
__global__ void k_cls(const float* __restrict__ f0, const float* __restrict__ f1,
                      const float* __restrict__ clsw, const float* __restrict__ clsb,
                      float* __restrict__ out) {
    int wave = threadIdx.x >> 5, lane = threadIdx.x & 31;
    int p = blockIdx.x * 8 + wave;
    if (p >= PP) return;
    int p2 = (p + PP - 1) % PP;   // f2 = roll(f1, 1)
    const float* a  = f0 + (size_t)p  * EMB;
    const float* b  = f1 + (size_t)p  * EMB;
    const float* b2 = f1 + (size_t)p2 * EMB;
    float sp = 0.f, sn = 0.f;
    for (int c = lane; c < EMB; c += 32) {
        float w = clsw[c], av = a[c];
        sp += fmaxf(av, b[c])  * w;
        sn += fmaxf(av, b2[c]) * w;
    }
    #pragma unroll
    for (int m = 16; m; m >>= 1) {
        sp += __shfl_xor(sp, m, 32);
        sn += __shfl_xor(sn, m, 32);
    }
    if (lane == 0) { out[p] = sp + clsb[0]; out[PP + p] = sn + clsb[0]; }
}

// =====================================================================
extern "C" void kernel_launch(void* const* d_in, const int* in_sizes, int n_in,
                              void* d_out, int out_size, void* d_ws, size_t ws_size,
                              hipStream_t stream) {
    const int*   x      = (const int*)d_in[0];
    const int*   ei     = (const int*)d_in[1];
    const int*   ea     = (const int*)d_in[2];
    const int*   fb     = (const int*)d_in[3];
    const int*   fei    = (const int*)d_in[4];
    const int*   dea    = (const int*)d_in[5];
    const float* aemb1  = (const float*)d_in[7];
    const float* aemb2  = (const float*)d_in[8];
    const float* gin_e1 = (const float*)d_in[9];
    const float* gin_e2 = (const float*)d_in[10];
    const float* gin_w1 = (const float*)d_in[11];
    const float* gin_b1 = (const float*)d_in[12];
    const float* gin_w2 = (const float*)d_in[13];
    const float* gin_b2 = (const float*)d_in[14];
    const float* bn_g   = (const float*)d_in[15];
    const float* bn_b   = (const float*)d_in[16];
    const float* pw1    = (const float*)d_in[17];
    const float* pb1    = (const float*)d_in[18];
    const float* pw2    = (const float*)d_in[19];
    const float* pb2    = (const float*)d_in[20];
    const float* gcw    = (const float*)d_in[21];
    const float* gcb    = (const float*)d_in[22];
    const float* gce1   = (const float*)d_in[23];
    const float* gce2   = (const float*)d_in[24];
    const float* clsw   = (const float*)d_in[25];
    const float* clsb   = (const float*)d_in[26];
    float* out = (float*)d_out;

    // ---- carve workspace ----
    char* wp = (char*)d_ws;
    auto carve = [&](size_t bytes) -> void* {
        void* r = (void*)wp;
        wp += (bytes + 255) & ~(size_t)255;
        return r;
    };
    float*  h     = (float*)carve((size_t)NN * EMB * 4);
    float*  agg   = (float*)carve((size_t)NN * EMB * 4);   // also reused as h2
    __bf16* aggb  = (__bf16*)carve((size_t)NN * PAD_E * 2);
    __bf16* z1    = (__bf16*)carve((size_t)NN * PAD_H * 2);
    __bf16* wtA   = (__bf16*)carve((size_t)PAD_H * PAD_H * 2);
    __bf16* wtB   = (__bf16*)carve((size_t)PAD_H * PAD_H * 2);
    float*  biasA = (float*)carve(PAD_H * 4);
    float*  biasB = (float*)carve(PAD_H * 4);
    float*  stats = (float*)carve(2 * EMB * 4);
    float*  ss    = (float*)carve(2 * EMB * 4);
    float*  seg   = (float*)carve((size_t)PP * EMB * 4);
    float*  cnt   = (float*)carve(PP * 4);
    __bf16* poolb = (__bf16*)carve((size_t)PP * PAD_E * 2);
    __bf16* prj1b = (__bf16*)carve((size_t)PP * PAD_E * 2);
    float*  outf  = (float*)carve((size_t)PP * EMB * 4);
    __bf16* outb  = (__bf16*)carve((size_t)PP * PAD_E * 2);
    float*  f0    = (float*)carve((size_t)PP * EMB * 4);
    float*  xw    = (float*)carve((size_t)PP * EMB * 4);
    float*  pred  = (float*)carve((size_t)PP * EMB * 4);
    float*  f1    = (float*)carve((size_t)PP * EMB * 4);
    float*  deg   = (float*)carve(PP * 4);
    float*  dis   = (float*)carve(PP * 4);

    const int T = 256;
    const int nodeBlocks = (int)(((size_t)NN * EMB + T - 1) / T);
    const int edgeBlocks = (int)(((size_t)EE * EMB + T - 1) / T);
    const int cvtBlocksN = (int)(((size_t)NN * (PAD_E / 2) + T - 1) / T);

    float* h2 = agg;   // GEMM2 output aliases agg (agg dead after cvt to aggb)

    // ---- encoder ----
    k_embed<<<nodeBlocks, T, 0, stream>>>(x, aemb1, aemb2, h);

    for (int l = 0; l < LAYERS; ++l) {
        const float* ge1 = gin_e1 + (size_t)l * 6 * EMB;
        const float* ge2 = gin_e2 + (size_t)l * 3 * EMB;
        k_agg_init<<<nodeBlocks, T, 0, stream>>>(h, ge1, ge2, agg);
        k_scatter<<<edgeBlocks, T, 0, stream>>>(h, ei, ea, ge1, ge2, agg);
        k_cvt_pad<<<cvtBlocksN, T, 0, stream>>>(agg, aggb, NN, EMB, PAD_E);

        // GEMM1: z1 = relu(aggb @ W1 + b1)  [N x 600], bf16 out
        k_conv_wT<<<(PAD_H * PAD_E + T - 1) / T, T, 0, stream>>>(
            gin_w1 + (size_t)l * EMB * HID, wtA, EMB, HID, PAD_E, PAD_H);
        k_pad_bias<<<(PAD_H + T - 1) / T, T, 0, stream>>>(
            gin_b1 + (size_t)l * HID, biasA, HID, PAD_H);
        {
            dim3 g((NN + 127) / 128, PAD_H / 128);
            gemm_wmma_bf16<true, false, true><<<g, T, 0, stream>>>(
                aggb, PAD_E, wtA, biasA, NN, HID, nullptr, z1, PAD_H);
        }
        // GEMM2: h2 = z1 @ W2 + b2  [N x 300], f32 out (aliases agg)
        k_conv_wT<<<(PAD_E * PAD_H + T - 1) / T, T, 0, stream>>>(
            gin_w2 + (size_t)l * HID * EMB, wtB, HID, EMB, PAD_H, PAD_E);
        k_pad_bias<<<(PAD_E + T - 1) / T, T, 0, stream>>>(
            gin_b2 + (size_t)l * EMB, biasB, EMB, PAD_E);
        {
            dim3 g((NN + 127) / 128, PAD_E / 128);
            gemm_wmma_bf16<false, true, false><<<g, T, 0, stream>>>(
                z1, PAD_H, wtB, biasB, NN, EMB, h2, nullptr, 0);
        }
        // BatchNorm (training stats) + relu (except last layer) -> h
        hipMemsetAsync(stats, 0, 2 * EMB * 4, stream);
        k_bn_stats<<<512, T, 0, stream>>>(h2, stats);
        k_bn_fin<<<(EMB + T - 1) / T, T, 0, stream>>>(
            stats, bn_g + (size_t)l * EMB, bn_b + (size_t)l * EMB, ss);
        k_bn_apply<<<nodeBlocks, T, 0, stream>>>(h2, ss, h, l < LAYERS - 1 ? 1 : 0);
    }

    // ---- fragment mean pool ----
    hipMemsetAsync(seg, 0, (size_t)PP * EMB * 4, stream);
    hipMemsetAsync(cnt, 0, PP * 4, stream);
    k_pool<<<nodeBlocks, T, 0, stream>>>(h, fb, seg, cnt);
    k_pool_div<<<(PP * PAD_E + T - 1) / T, T, 0, stream>>>(seg, cnt, poolb);

    // ---- projector MLP ----
    k_conv_wT<<<(PAD_E * PAD_E + T - 1) / T, T, 0, stream>>>(pw1, wtA, EMB, EMB, PAD_E, PAD_E);
    k_pad_bias<<<(PAD_E + T - 1) / T, T, 0, stream>>>(pb1, biasA, EMB, PAD_E);
    {
        dim3 g((PP + 127) / 128, PAD_E / 128);
        gemm_wmma_bf16<true, false, true><<<g, T, 0, stream>>>(
            poolb, PAD_E, wtA, biasA, PP, EMB, nullptr, prj1b, PAD_E);
    }
    k_conv_wT<<<(PAD_E * PAD_E + T - 1) / T, T, 0, stream>>>(pw2, wtB, EMB, EMB, PAD_E, PAD_E);
    k_pad_bias<<<(PAD_E + T - 1) / T, T, 0, stream>>>(pb2, biasB, EMB, PAD_E);
    {
        dim3 g((PP + 127) / 128, PAD_E / 128);
        gemm_wmma_bf16<false, true, true><<<g, T, 0, stream>>>(
            prj1b, PAD_E, wtB, biasB, PP, EMB, outf, outb, PAD_E);
    }
    k_rownorm<<<(PP + 7) / 8, T, 0, stream>>>(outf, f0, PP);

    // ---- GCN predictor ----
    k_conv_wT<<<(PAD_E * PAD_E + T - 1) / T, T, 0, stream>>>(gcw, wtA, EMB, EMB, PAD_E, PAD_E);
    k_pad_bias<<<(PAD_E + T - 1) / T, T, 0, stream>>>(gcb, biasA, EMB, PAD_E);
    {
        dim3 g((PP + 127) / 128, PAD_E / 128);
        gemm_wmma_bf16<false, true, false><<<g, T, 0, stream>>>(
            outb, PAD_E, wtA, biasA, PP, EMB, xw, nullptr, 0);
    }
    hipMemsetAsync(deg, 0, PP * 4, stream);
    k_deg<<<(2 * FE2 + T - 1) / T, T, 0, stream>>>(fei, deg);
    k_dis<<<(PP + T - 1) / T, T, 0, stream>>>(deg, dis);
    hipMemsetAsync(pred, 0, (size_t)PP * EMB * 4, stream);
    {
        size_t tot = (size_t)2 * FE2 * EMB;
        k_gcn_scatter<<<(int)((tot + T - 1) / T), T, 0, stream>>>(
            xw, fei, dea, gce1, gce2, dis, pred);
    }
    k_rownorm<<<(PP + 7) / 8, T, 0, stream>>>(pred, f1, PP);

    // ---- classifier ----
    k_cls<<<(PP + 7) / 8, T, 0, stream>>>(f0, f1, clsw, clsb, out);

    (void)in_sizes; (void)n_in; (void)out_size; (void)ws_size;
}
